// MultiHeadTransformerBlock_33440615367402
// MI455X (gfx1250) — matrix operand order repaired and checked by reading, not compiled, not run on hardware
//
#include <hip/hip_runtime.h>
#include <hip/hip_bf16.h>

// ---------------------------------------------------------------------------
// MI455X (gfx1250) transformer block, bf16 WMMA everywhere, fp32 accumulate.
// GEMMs: 128x128 block tile, K-step 32, 8 waves/WG, DOUBLE-BUFFERED LDS with
// gfx1250 async global->LDS copies (ASYNCcnt) for A tiles and software-
// pipelined register staging for (transposed) B tiles.
// Attention: flash-style causal, per-wave 16-row Q tile, WMMA for S and PV,
// async-staged K tile, transposed V tile.
// ---------------------------------------------------------------------------

typedef __attribute__((ext_vector_type(8)))  float           v8f;
typedef __attribute__((ext_vector_type(16))) __bf16          v16bf;
typedef __attribute__((ext_vector_type(16))) unsigned short  v16us;
typedef __attribute__((ext_vector_type(4)))  int             v4i;

#if defined(__gfx1250__) && __has_builtin(__builtin_amdgcn_global_load_async_to_lds_b128)
#define ATB_ASYNC 1
typedef __attribute__((address_space(1))) v4i v4i_g;   // global
typedef __attribute__((address_space(3))) v4i v4i_l;   // LDS
#else
#define ATB_ASYNC 0
#endif

static __device__ __forceinline__ void copy16_to_lds(const unsigned short* g,
                                                     unsigned short* l) {
#if ATB_ASYNC
    __builtin_amdgcn_global_load_async_to_lds_b128((v4i_g*)g, (v4i_l*)l, 0, 0);
#else
    *reinterpret_cast<uint4*>(l) = *reinterpret_cast<const uint4*>(g);
#endif
}

static __device__ __forceinline__ void wait_lds_copies() {
#if ATB_ASYNC
#if __has_builtin(__builtin_amdgcn_s_wait_asynccnt)
    __builtin_amdgcn_s_wait_asynccnt(0);
#else
    asm volatile("s_wait_asynccnt 0" ::: "memory");
#endif
#endif
}

static __device__ __forceinline__ v16bf as_bf(v16us u) {
    return __builtin_bit_cast(v16bf, u);
}

static __device__ __forceinline__ unsigned short f32_to_bf16(float f) {
    unsigned int u = __float_as_uint(f);
    unsigned int r = (u + 0x7FFFu + ((u >> 16) & 1u)) >> 16;   // RNE
    return (unsigned short)r;
}

static __device__ __forceinline__ v8f vzero8() {
    v8f r;
#pragma unroll
    for (int i = 0; i < 8; ++i) r[i] = 0.0f;
    return r;
}

// ---------------------------------------------------------------------------
// Elementwise fp32 -> bf16
// ---------------------------------------------------------------------------
__global__ __launch_bounds__(256) void cvt_bf16_kernel(const float* __restrict__ in,
                                                       unsigned short* __restrict__ out,
                                                       int n) {
    int i = blockIdx.x * 256 + threadIdx.x;
    if (i < n) out[i] = f32_to_bf16(in[i]);
}

// ---------------------------------------------------------------------------
// RMSNorm over rows of length 1024; output bf16. One block per row.
// ---------------------------------------------------------------------------
__global__ __launch_bounds__(256) void rmsnorm_kernel(const float* __restrict__ X,
                                                      const float* __restrict__ gamma,
                                                      unsigned short* __restrict__ Y) {
    __shared__ float red[256];
    const int tid = threadIdx.x;
    const size_t row = blockIdx.x;
    const float* x = X + row * 1024;

    float v[4];
    float s = 0.0f;
#pragma unroll
    for (int k = 0; k < 4; ++k) {
        v[k] = x[tid + 256 * k];
        s += v[k] * v[k];
    }
    red[tid] = s;
    __syncthreads();
#pragma unroll
    for (int off = 128; off > 0; off >>= 1) {
        if (tid < off) red[tid] += red[tid + off];
        __syncthreads();
    }
    const float r = rsqrtf(red[0] * (1.0f / 1024.0f) + 1e-6f);
#pragma unroll
    for (int k = 0; k < 4; ++k) {
        int c = tid + 256 * k;
        Y[row * 1024 + c] = f32_to_bf16(v[k] * r * gamma[c]);
    }
}

// ---------------------------------------------------------------------------
// bf16 GEMM: C(MxN) = A(MxK) * B(KxN), fp32 accumulate via WMMA 16x16x32.
// Double-buffered LDS; A staged by async global->LDS copies, B staged via
// one-iteration-ahead register pipeline + transpose scatter into LDS.
// MODE 0: store bf16; MODE 1: relu -> bf16; MODE 2: fp32 out = resid + acc.
// ---------------------------------------------------------------------------
template <int MODE>
__global__ __launch_bounds__(256) void gemm_bf16_kernel(
    const unsigned short* __restrict__ A, const unsigned short* __restrict__ B,
    unsigned short* __restrict__ Cb, float* __restrict__ Cf,
    const float* __restrict__ resid, int M, int N, int K) {

    __shared__ unsigned short lA[2][128][40];    // A tile 128x32 (+pad), x2 buffers
    __shared__ unsigned short lBt[2][128][40];   // B tile TRANSPOSED [n][k], x2

    const int tid  = threadIdx.x;
    const int lane = tid & 31;
    const int wave = tid >> 5;
    const int half = (lane >> 4) & 1;
    const int ln   = lane & 15;
    const int wm   = wave >> 1;              // 0..3
    const int wn   = wave & 1;               // 0..1
    const int mbase = blockIdx.y * 128;
    const int nbase = blockIdx.x * 128;
    const int wrow  = wm * 32;
    const int wcol  = wn * 64;

    // staging coordinates (2 chunks of 8 bf16 per thread per tile)
    const int a_r0 = tid >> 2,  a_c = (tid & 3) << 3;     // A rows a_r0, a_r0+64
    const int b_r0 = tid >> 4,  b_c = (tid & 15) << 3;    // B rows b_r0, b_r0+16

    v8f acc[2][4];
#pragma unroll
    for (int mi = 0; mi < 2; ++mi)
#pragma unroll
        for (int ni = 0; ni < 4; ++ni) acc[mi][ni] = vzero8();

    // ---- prologue: pipeline B(kk=0) into regs, async A(kk=0) into buf 0 ----
    uint4 breg[2];
#pragma unroll
    for (int ci = 0; ci < 2; ++ci)
        breg[ci] = *reinterpret_cast<const uint4*>(
            &B[(size_t)(b_r0 + 16 * ci) * N + nbase + b_c]);
#pragma unroll
    for (int ci = 0; ci < 2; ++ci) {
        const int r = a_r0 + 64 * ci;
        copy16_to_lds(&A[(size_t)(mbase + r) * K + a_c], &lA[0][r][a_c]);
    }

    const int ksteps = K >> 5;
    for (int kki = 0; kki < ksteps; ++kki) {
        const int kk  = kki << 5;
        const int buf = kki & 1;
        const bool more = (kki + 1 < ksteps);

        // Scatter pipelined B regs (transpose) into this buffer.
#pragma unroll
        for (int ci = 0; ci < 2; ++ci) {
            const unsigned short* bs = reinterpret_cast<const unsigned short*>(&breg[ci]);
            const int r = b_r0 + 16 * ci;
#pragma unroll
            for (int j = 0; j < 8; ++j) lBt[buf][b_c + j][r] = bs[j];
        }
        // Pipeline next B tile into regs (latency hidden behind compute).
        if (more) {
#pragma unroll
            for (int ci = 0; ci < 2; ++ci)
                breg[ci] = *reinterpret_cast<const uint4*>(
                    &B[(size_t)(kk + 32 + b_r0 + 16 * ci) * N + nbase + b_c]);
        }

        wait_lds_copies();          // our async A copies for this buffer done
        __syncthreads();            // everyone's A + B staging visible

        // Kick off async A copies for the NEXT buffer; overlaps with compute.
        if (more) {
#pragma unroll
            for (int ci = 0; ci < 2; ++ci) {
                const int r = a_r0 + 64 * ci;
                copy16_to_lds(&A[(size_t)(mbase + r) * K + kk + 32 + a_c],
                              &lA[buf ^ 1][r][a_c]);
            }
        }

        // A fragments (ISA 16-bit A layout): two 16B contiguous groups per lane.
        v16bf afrag[2];
#pragma unroll
        for (int mi = 0; mi < 2; ++mi) {
            v16us u;
            const int r = wrow + mi * 16 + ln;
#pragma unroll
            for (int j = 0; j < 8; ++j) {
                u[j]     = lA[buf][r][8 * half + j];
                u[8 + j] = lA[buf][r][16 + 8 * half + j];
            }
            afrag[mi] = as_bf(u);
        }
        // B fragments from transposed tile: two 16B contiguous groups per lane.
        v16bf bfrag[4];
#pragma unroll
        for (int ni = 0; ni < 4; ++ni) {
            v16us u;
            const int c = wcol + ni * 16 + ln;
#pragma unroll
            for (int i = 0; i < 8; ++i) { u[i] = lBt[buf][c][8 * half + i]; u[8 + i] = lBt[buf][c][16 + 8 * half + i]; }
            bfrag[ni] = as_bf(u);
        }
#pragma unroll
        for (int mi = 0; mi < 2; ++mi)
#pragma unroll
            for (int ni = 0; ni < 4; ++ni)
                acc[mi][ni] = __builtin_amdgcn_wmma_f32_16x16x32_bf16(
                    false, afrag[mi], false, bfrag[ni], (short)0, acc[mi][ni],
                    false, false);
        // No trailing barrier: next iteration writes the OTHER buffer; reuse of
        // this buffer only happens after the next iteration's barrier.
    }

    // Epilogue. C layout: row = i + 8*half, col = ln.  Straight-line per MODE.
#pragma unroll
    for (int mi = 0; mi < 2; ++mi)
#pragma unroll
        for (int ni = 0; ni < 4; ++ni)
#pragma unroll
            for (int i = 0; i < 8; ++i) {
                const int r = mbase + wrow + mi * 16 + i + 8 * half;
                const int c = nbase + wcol + ni * 16 + ln;
                const float v = acc[mi][ni][i];
                if constexpr (MODE == 0) {
                    Cb[(size_t)r * N + c] = f32_to_bf16(v);
                } else if constexpr (MODE == 1) {
                    Cb[(size_t)r * N + c] = f32_to_bf16(v > 0.0f ? v : 0.0f);
                } else {
                    Cf[(size_t)r * N + c] = resid[(size_t)r * N + c] + v;
                }
            }
}

// ---------------------------------------------------------------------------
// Causal flash attention. QKV bf16 (b*t, 3072): Q [0,1024), K [1024,2048),
// V [2048,3072). Grid (t/64, h, b); 128 threads = 4 waves; wave owns a 16-row
// Q tile; WG stages K (async, row-major) and V (transposed) 32x64 tiles.
// Output O bf16 (b*t, 1024) in (b,t,h,d) layout (== concat heads).
// ---------------------------------------------------------------------------
__global__ __launch_bounds__(128) void attn_kernel(
    const unsigned short* __restrict__ QKV, unsigned short* __restrict__ O,
    int T) {

    __shared__ unsigned short kt[32][72];       // K tile: [kpos][d]
    __shared__ unsigned short vtT[64][40];      // V tile TRANSPOSED: [d][kpos]
    __shared__ unsigned short ps[4][16][40];    // per-wave P staging 16x32

    const int tid  = threadIdx.x;
    const int lane = tid & 31;
    const int wave = tid >> 5;
    const int half = (lane >> 4) & 1;
    const int ln   = lane & 15;
    const int qb   = blockIdx.x;
    const int head = blockIdx.y;
    const int bz   = blockIdx.z;
    const int row0 = qb * 64 + wave * 16;        // token row of this wave's Q tile
    const size_t base = (size_t)bz * T * 3072;

    // Q A-fragments for d 0..31 and 32..63 (loaded once, from global)
    v16bf aq[2];
#pragma unroll
    for (int kc = 0; kc < 2; ++kc) {
        v16us u;
        const size_t ro = base + (size_t)(row0 + ln) * 3072 + head * 64 + kc * 32;
#pragma unroll
        for (int j = 0; j < 8; ++j) {
            u[j]     = QKV[ro + 8 * half + j];
            u[8 + j] = QKV[ro + 16 + 8 * half + j];
        }
        aq[kc] = as_bf(u);
    }

    v8f oacc[4];
#pragma unroll
    for (int ni = 0; ni < 4; ++ni) oacc[ni] = vzero8();
    float mrun[8], lrun[8];
#pragma unroll
    for (int i = 0; i < 8; ++i) { mrun[i] = -3.0e38f; lrun[i] = 0.0f; }

    const int ktiles = qb * 2 + 2;               // covers kpos <= qb*64+63
    for (int kti = 0; kti < ktiles; ++kti) {
        const int k0 = kti * 32;

        // Cooperative staging: K via async global->LDS copy, V transposed.
#pragma unroll
        for (int ci = 0; ci < 2; ++ci) {
            int ch = tid + 128 * ci;
            int r  = ch >> 3;
            int cc = (ch & 7) << 3;
            const size_t g = base + (size_t)(k0 + r) * 3072 + head * 64 + cc;
            copy16_to_lds(&QKV[g + 1024], &kt[r][cc]);
            uint4 vv = *reinterpret_cast<const uint4*>(&QKV[g + 2048]);
            const unsigned short* vs = reinterpret_cast<const unsigned short*>(&vv);
#pragma unroll
            for (int j = 0; j < 8; ++j) vtT[cc + j][r] = vs[j];
        }
        // Prefetch next V tile (global_prefetch_b8).
        if (kti + 1 < ktiles) {
            const size_t gn = base + (size_t)(k0 + 32 + (tid >> 3)) * 3072 +
                              head * 64 + ((tid & 7) << 3);
            __builtin_prefetch(&QKV[gn + 2048], 0, 3);
        }
        wait_lds_copies();
        __syncthreads();

        // S tile 16x32 = two 16x16 wmma outputs, each chaining K=64 (2 calls).
        v8f s[2];
        s[0] = vzero8(); s[1] = vzero8();
#pragma unroll
        for (int nt = 0; nt < 2; ++nt) {
#pragma unroll
            for (int kc = 0; kc < 2; ++kc) {
                v16us u;                          // B[d][n] = K[kpos=n][d]; contiguous in d
#pragma unroll
                for (int i = 0; i < 8; ++i) {
                    u[i]     = kt[nt * 16 + ln][kc * 32 + 8 * half + i];
                    u[8 + i] = kt[nt * 16 + ln][kc * 32 + 16 + 8 * half + i];
                }
                s[nt] = __builtin_amdgcn_wmma_f32_16x16x32_bf16(
                    false, aq[kc], false, as_bf(u), (short)0, s[nt], false, false);
            }
        }

        // Scale + causal mask + per-row tile max.
        float mt[8];
#pragma unroll
        for (int i = 0; i < 8; ++i) {
            const int qpos = row0 + i + 8 * half;
            const int kp0  = k0 + ln;
            const int kp1  = k0 + 16 + ln;
            float a0 = (kp0 <= qpos) ? s[0][i] * 0.125f : -3.0e38f;
            float a1 = (kp1 <= qpos) ? s[1][i] * 0.125f : -3.0e38f;
            s[0][i] = a0; s[1][i] = a1;
            mt[i] = fmaxf(a0, a1);
        }
#pragma unroll
        for (int m = 1; m < 16; m <<= 1)
#pragma unroll
            for (int i = 0; i < 8; ++i)
                mt[i] = fmaxf(mt[i], __shfl_xor(mt[i], m, 32));

        // Flash rescale + exp.
        float rs[8];
#pragma unroll
        for (int i = 0; i < 8; ++i) {
            const float mn    = fmaxf(mrun[i], mt[i]);
            const float alpha = __expf(mrun[i] - mn);
            mrun[i] = mn;
            const float p0 = __expf(s[0][i] - mn);
            const float p1 = __expf(s[1][i] - mn);
            s[0][i] = p0; s[1][i] = p1;
            rs[i]   = p0 + p1;
            lrun[i] *= alpha;
#pragma unroll
            for (int ni = 0; ni < 4; ++ni) oacc[ni][i] *= alpha;
        }
#pragma unroll
        for (int m = 1; m < 16; m <<= 1)
#pragma unroll
            for (int i = 0; i < 8; ++i)
                rs[i] += __shfl_xor(rs[i], m, 32);
#pragma unroll
        for (int i = 0; i < 8; ++i) lrun[i] += rs[i];

        // Re-layout P (C layout -> A fragment) via per-wave LDS staging.
#pragma unroll
        for (int nt = 0; nt < 2; ++nt)
#pragma unroll
            for (int i = 0; i < 8; ++i)
                ps[wave][i + 8 * half][nt * 16 + ln] = f32_to_bf16(s[nt][i]);
        // Same-wave LDS ops are in-order (ISA 7.3): read-after-write is safe.
        v16us up;
#pragma unroll
        for (int j = 0; j < 8; ++j) {
            up[j]     = ps[wave][ln][8 * half + j];
            up[8 + j] = ps[wave][ln][16 + 8 * half + j];
        }
        const v16bf ap = as_bf(up);

        // O(16x64) += P(16x32) * V(32x64); V fragments contiguous from vtT.
#pragma unroll
        for (int ni = 0; ni < 4; ++ni) {
            v16us ub;                              // B[kpos][d=n] = vtT[d][kpos]
#pragma unroll
            for (int i = 0; i < 8; ++i) {
                ub[i]     = vtT[ni * 16 + ln][8 * half + i];
                ub[8 + i] = vtT[ni * 16 + ln][16 + 8 * half + i];
            }
            oacc[ni] = __builtin_amdgcn_wmma_f32_16x16x32_bf16(
                false, ap, false, as_bf(ub), (short)0, oacc[ni], false, false);
        }
        __syncthreads();
    }

    // Normalize by running sum, store bf16 in (b,t,h*dk) layout.
#pragma unroll
    for (int ni = 0; ni < 4; ++ni)
#pragma unroll
        for (int i = 0; i < 8; ++i) {
            const int row = row0 + i + 8 * half;
            const int col = head * 64 + ni * 16 + ln;
            const float v = oacc[ni][i] / lrun[i];
            O[(size_t)bz * T * 1024 + (size_t)row * 1024 + col] = f32_to_bf16(v);
        }
}

// ---------------------------------------------------------------------------
// Host-side orchestration
// ---------------------------------------------------------------------------
extern "C" void kernel_launch(void* const* d_in, const int* in_sizes, int n_in,
                              void* d_out, int out_size, void* d_ws, size_t ws_size,
                              hipStream_t stream) {
    (void)in_sizes; (void)n_in; (void)out_size; (void)ws_size;

    const float* X       = (const float*)d_in[0];   // (2,2048,1024)
    const float* W_QKV   = (const float*)d_in[1];   // (1024,3072)
    const float* W_O     = (const float*)d_in[2];   // (1024,1024)
    const float* Wff_in  = (const float*)d_in[3];   // (1024,4096)
    const float* Wff_out = (const float*)d_in[4];   // (4096,1024)
    const float* gamma1  = (const float*)d_in[5];
    const float* gamma2  = (const float*)d_in[6];

    const int B = 2, T = 2048, D = 1024, H3 = 3072, FF = 4096;
    const int M = B * T;                            // 4096 token rows

    char* ws = (char*)d_ws;
    size_t off = 0;
    auto give = [&](size_t bytes) -> char* {
        char* p = ws + off;
        off = (off + bytes + 255) & ~(size_t)255;
        return p;
    };
    unsigned short* Wqkv_b = (unsigned short*)give((size_t)D * H3 * 2);
    unsigned short* Wo_b   = (unsigned short*)give((size_t)D * D * 2);
    unsigned short* Wfi_b  = (unsigned short*)give((size_t)D * FF * 2);
    unsigned short* Wfo_b  = (unsigned short*)give((size_t)FF * D * 2);
    unsigned short* Xn1    = (unsigned short*)give((size_t)M * D * 2);
    unsigned short* QKVb   = (unsigned short*)give((size_t)M * H3 * 2);
    unsigned short* AttnO  = (unsigned short*)give((size_t)M * D * 2);
    float*          X1     = (float*)give((size_t)M * D * 4);
    unsigned short* Xn2    = (unsigned short*)give((size_t)M * D * 2);
    unsigned short* Hbuf   = (unsigned short*)give((size_t)M * FF * 2);

    // 1) bf16 weight copies
    {
        int n;
        n = D * H3; cvt_bf16_kernel<<<(n + 255) / 256, 256, 0, stream>>>(W_QKV,   Wqkv_b, n);
        n = D * D;  cvt_bf16_kernel<<<(n + 255) / 256, 256, 0, stream>>>(W_O,     Wo_b,   n);
        n = D * FF; cvt_bf16_kernel<<<(n + 255) / 256, 256, 0, stream>>>(Wff_in,  Wfi_b,  n);
        n = FF * D; cvt_bf16_kernel<<<(n + 255) / 256, 256, 0, stream>>>(Wff_out, Wfo_b,  n);
    }

    // 2) Xn1 = rmsnorm(X, gamma1)  (bf16)
    rmsnorm_kernel<<<M, 256, 0, stream>>>(X, gamma1, Xn1);

    // 3) QKV = Xn1 @ W_QKV  (bf16 out)
    gemm_bf16_kernel<0><<<dim3(H3 / 128, M / 128), 256, 0, stream>>>(
        Xn1, Wqkv_b, QKVb, nullptr, nullptr, M, H3, D);

    // 4) causal attention -> AttnO (bf16, heads concatenated)
    attn_kernel<<<dim3(T / 64, 16, B), 128, 0, stream>>>(QKVb, AttnO, T);

    // 5) X1 = X + AttnO @ W_O  (fp32 out)
    gemm_bf16_kernel<2><<<dim3(D / 128, M / 128), 256, 0, stream>>>(
        AttnO, Wo_b, nullptr, X1, X, M, D, D);

    // 6) Xn2 = rmsnorm(X1, gamma2)
    rmsnorm_kernel<<<M, 256, 0, stream>>>(X1, gamma2, Xn2);

    // 7) H = relu(Xn2 @ Wff_in)  (bf16 out)
    gemm_bf16_kernel<1><<<dim3(FF / 128, M / 128), 256, 0, stream>>>(
        Xn2, Wfi_b, Hbuf, nullptr, nullptr, M, FF, D);

    // 8) out = X1 + H @ Wff_out  (fp32 out)
    gemm_bf16_kernel<2><<<dim3(D / 128, M / 128), 256, 0, stream>>>(
        Hbuf, Wfo_b, nullptr, (float*)d_out, X1, M, D, FF);
}
